// Modulated2d_28690381538077
// MI455X (gfx1250) — matrix operand order repaired and checked by reading, not compiled, not run on hardware
//
#include <hip/hip_runtime.h>

typedef __attribute__((ext_vector_type(16))) _Float16 v16h;
typedef __attribute__((ext_vector_type(8)))  _Float16 v8h;
typedef __attribute__((ext_vector_type(8)))  float    v8f;

#define BATCH 32
#define CIN   256
#define COUT  256
#define HW    64
#define TAPS  9
#define KDIM  (TAPS * CIN)          // 2304, K order = [tap][cin]

#define LDS_CPAD 40                 // cin slot padded 32 -> 40 halves (80 B)
#define LDS_COLS 66                 // 64 cols + zero halo both sides
#define TILE_H   (3 * LDS_COLS * LDS_CPAD)   // 7920 halves per buffer

union ABu { v16h v; v8h h[2]; };

// ---- CDNA5 async global->LDS copy (16 bytes per lane), ASYNCcnt tracked ----
__device__ __forceinline__ void copy16_async(const void* g, void* l) {
#if __has_builtin(__builtin_amdgcn_load_to_lds)
    __builtin_amdgcn_load_to_lds(
        (void*)g,
        (__attribute__((address_space(3))) void*)(unsigned)(size_t)l,
        16, 0, 0);
#else
    unsigned lo = (unsigned)(size_t)l;
    asm volatile("global_load_async_to_lds_b128 %0, %1, off"
                 :: "v"(lo), "v"(g) : "memory");
#endif
}

__device__ __forceinline__ void wait_async0() {
#if __has_builtin(__builtin_amdgcn_s_wait_asynccnt)
    __builtin_amdgcn_s_wait_asynccnt(0);
#else
    asm volatile("s_wait_asynccnt 0" ::: "memory");
#endif
}

// ---------------------------------------------------------------------------
// Kernel 0: x [B,CIN,H,W] f32  ->  xh [B,H,W,CIN] f16  (transpose + convert)
// ---------------------------------------------------------------------------
__global__ __launch_bounds__(256) void xcast_kernel(
    const float* __restrict__ x, _Float16* __restrict__ xh)
{
    const int gy  = blockIdx.x;          // image row
    const int b   = blockIdx.y;
    const int cin = threadIdx.x;         // 256 == CIN
    const float* src = x + ((size_t)(b * CIN + cin)) * (HW * HW) + gy * HW;
    _Float16*    dst = xh + ((size_t)(b * HW + gy)) * HW * CIN + cin;
#pragma unroll 8
    for (int col = 0; col < HW; ++col)
        dst[col * CIN] = (_Float16)src[col];
}

// ---------------------------------------------------------------------------
// Kernel 1: per-(b,cout) modulation + demodulation -> f16  [B,COUT,TAPS,CIN]
// ---------------------------------------------------------------------------
__global__ __launch_bounds__(256) void modulate_kernel(
    const float* __restrict__ y,
    const float* __restrict__ weight,
    _Float16* __restrict__ wmod)
{
    const int cout = blockIdx.x;
    const int b    = blockIdx.y;
    const int cin  = threadIdx.x;

    const float s = y[b * CIN + cin] + 1.0f;

    float wl[TAPS];
    float ss = 0.0f;
    const float* wp = weight + (size_t)cout * KDIM + (size_t)cin * TAPS;
#pragma unroll
    for (int j = 0; j < TAPS; ++j) {
        float v = wp[j] * s;
        wl[j] = v;
        ss += v * v;
    }

    __shared__ float red[256];
    red[cin] = ss;
    __syncthreads();
#pragma unroll
    for (int off = 128; off > 0; off >>= 1) {
        if (cin < off) red[cin] += red[cin + off];
        __syncthreads();
    }
    const float d = rsqrtf(red[0] + 1e-8f);

    _Float16* dst = wmod + (((size_t)b * COUT + cout) * TAPS) * CIN + cin;
#pragma unroll
    for (int j = 0; j < TAPS; ++j)
        dst[j * CIN] = (_Float16)(wl[j] * d);
}

// ---------------------------------------------------------------------------
// Kernel 2: implicit-GEMM conv, v_wmma_f32_16x16x32_f16.
// Workgroup tile = 128 couts x 64 pixels (one image row) for one sample.
// 8 waves: wave w -> cout group w (16 couts), 4 pixel groups (4 accumulators,
// A fragment reused 4x). Double-buffered async-DMA LDS staging of x.
// ---------------------------------------------------------------------------
__global__ __launch_bounds__(256) void conv_wmma_kernel(
    const _Float16* __restrict__ xh,     // [B, H, W, CIN]
    const _Float16* __restrict__ wmod,   // [B, COUT, TAPS, CIN]
    const float*    __restrict__ bias,
    float*          __restrict__ out)    // [B, COUT, H, W]
{
    const int yrow     = blockIdx.x;          // 0..63
    const int coutTile = blockIdx.y;          // 0..1 (128 couts each)
    const int b        = blockIdx.z;
    const int tid      = threadIdx.x;
    const int lane     = tid & 31;
    const int wave     = tid >> 5;            // 0..7 == cout group

    __shared__ __align__(16) _Float16 bt[2][TILE_H];

    // ---- zero both buffers once (halo + out-of-range rows stay zero) ----
    {
        v8h z = {};
        _Float16* base = &bt[0][0];
        for (int i = tid; i < (2 * TILE_H) / 8; i += 256)
            *(v8h*)(base + i * 8) = z;
    }

    // ---- per-thread DMA slots: j=0..2 -> (row j, col tid>>2, 16B seg tid&3)
    const int seg  = tid & 3;
    const int colS = tid >> 2;
    const char* xbase = (const char*)(xh + (size_t)b * HW * HW * CIN);
    unsigned gOff[3], lOff[3];
    bool     valid[3];
#pragma unroll
    for (int j = 0; j < 3; ++j) {
        int gy   = yrow - 1 + j;
        valid[j] = (gy >= 0) && (gy < HW);
        gOff[j]  = valid[j] ? (unsigned)((gy * HW + colS) * (CIN * 2) + seg * 16) : 0u;
        lOff[j]  = (unsigned)(((j * LDS_COLS + colS + 1) * LDS_CPAD) * 2 + seg * 16);
    }

    __syncthreads();   // zero-fill visible before any DMA writes land

    // ---- wave tile assignment ----
    const int ncol  = lane & 15;
    const int khalf = lane >> 4;

    const _Float16* arowBase =
        wmod + ((size_t)b * COUT + coutTile * 128 + wave * 16 + ncol) * KDIM + khalf * 16;

    v8f acc0 = {}, acc1 = {}, acc2 = {}, acc3 = {};

    // prologue: DMA chunk 0 into buffer 0
#pragma unroll
    for (int j = 0; j < 3; ++j)
        if (valid[j]) copy16_async(xbase + gOff[j], (char*)&bt[0][0] + lOff[j]);

#pragma unroll 1
    for (int c = 0; c < 8; ++c) {
        wait_async0();
        __syncthreads();                       // buffer c&1 ready for all waves

        if (c < 7) {                           // DMA next chunk into other buffer
            const char* src = xbase + (c + 1) * 32 * 2;   // cin0 byte offset
            char* dstb = (char*)&bt[(c + 1) & 1][0];
#pragma unroll
            for (int j = 0; j < 3; ++j)
                if (valid[j]) copy16_async(src + gOff[j], dstb + lOff[j]);
        }

        const _Float16* ap  = arowBase + c * 32;
        const _Float16* bp0 = &bt[c & 1][ncol * LDS_CPAD + khalf * 16];

#pragma unroll
        for (int tap = 0; tap < TAPS; ++tap) {
            const int kh = tap / 3;
            const int kw = tap % 3;
            const int bo = (kh * LDS_COLS + kw) * LDS_CPAD;   // immediate

            ABu a;
            a.h[0] = *(const v8h*)(ap + tap * CIN);
            a.h[1] = *(const v8h*)(ap + tap * CIN + 8);

            ABu b0, b1, b2, b3;
            b0.h[0] = *(const v8h*)(bp0 + bo);
            b0.h[1] = *(const v8h*)(bp0 + bo + 8);
            b1.h[0] = *(const v8h*)(bp0 + bo + 16 * LDS_CPAD);
            b1.h[1] = *(const v8h*)(bp0 + bo + 16 * LDS_CPAD + 8);
            b2.h[0] = *(const v8h*)(bp0 + bo + 32 * LDS_CPAD);
            b2.h[1] = *(const v8h*)(bp0 + bo + 32 * LDS_CPAD + 8);
            b3.h[0] = *(const v8h*)(bp0 + bo + 48 * LDS_CPAD);
            b3.h[1] = *(const v8h*)(bp0 + bo + 48 * LDS_CPAD + 8);

            acc0 = __builtin_amdgcn_wmma_f32_16x16x32_f16(
                false, a.v, false, b0.v, (short)0, acc0, false, false);
            acc1 = __builtin_amdgcn_wmma_f32_16x16x32_f16(
                false, a.v, false, b1.v, (short)0, acc1, false, false);
            acc2 = __builtin_amdgcn_wmma_f32_16x16x32_f16(
                false, a.v, false, b2.v, (short)0, acc2, false, false);
            acc3 = __builtin_amdgcn_wmma_f32_16x16x32_f16(
                false, a.v, false, b3.v, (short)0, acc3, false, false);
        }
    }

    // ---- epilogue: D layout m = r + khalf*8, n = lane&15; bias vectorized
    const int cob = coutTile * 128 + wave * 16 + khalf * 8;
    float4 bv0 = *(const float4*)(bias + cob);
    float4 bv1 = *(const float4*)(bias + cob + 4);
    float bv[8] = {bv0.x, bv0.y, bv0.z, bv0.w, bv1.x, bv1.y, bv1.z, bv1.w};

#pragma unroll
    for (int r = 0; r < 8; ++r) {
        size_t obase = (((size_t)b * COUT + cob + r) * HW + yrow) * HW + ncol;
        out[obase]      = acc0[r] + bv[r];
        out[obase + 16] = acc1[r] + bv[r];
        out[obase + 32] = acc2[r] + bv[r];
        out[obase + 48] = acc3[r] + bv[r];
    }
}

// ---------------------------------------------------------------------------
extern "C" void kernel_launch(void* const* d_in, const int* in_sizes, int n_in,
                              void* d_out, int out_size, void* d_ws, size_t ws_size,
                              hipStream_t stream) {
    const float* x      = (const float*)d_in[0];   // [32,256,64,64]
    const float* y      = (const float*)d_in[1];   // [32,256]
    const float* weight = (const float*)d_in[2];   // [256,256,3,3]
    const float* bias   = (const float*)d_in[3];   // [256]
    float* out          = (float*)d_out;

    _Float16* xh   = (_Float16*)d_ws;                          // 67.1 MB
    _Float16* wmod = xh + (size_t)BATCH * HW * HW * CIN;       // +37.7 MB

    dim3 g0(HW, BATCH);
    xcast_kernel<<<g0, 256, 0, stream>>>(x, xh);

    dim3 g1(COUT, BATCH);
    modulate_kernel<<<g1, 256, 0, stream>>>(y, weight, wmod);

    dim3 g2(HW, COUT / 128, BATCH);
    conv_wmma_kernel<<<g2, 256, 0, stream>>>(xh, wmod, bias, out);
}
